// AiVadModel_38414187495448
// MI455X (gfx1250) — matrix operand order repaired and checked by reading, not compiled, not run on hardware
//
#include <hip/hip_runtime.h>
#include <hip/hip_bf16.h>

typedef __attribute__((ext_vector_type(16))) __bf16 v16bf;
typedef __attribute__((ext_vector_type(8)))  float  v8f;

#define LN2PI_F 1.8378770664093453f
#define FLT_BIG 3.402823466e38f

// ---------------------------------------------------------------------------
// Velocity GMM negative log-likelihood (K=5, D=8): one thread per sample.
// ---------------------------------------------------------------------------
__global__ void vel_gmm_kernel(const float* __restrict__ x,      // [N,8]
                               const float* __restrict__ means,  // [5,8]
                               const float* __restrict__ prec,   // [5,8,8]
                               const float* __restrict__ logw,   // [5]
                               const float* __restrict__ vmin,
                               const float* __restrict__ vmax,
                               float* __restrict__ out,          // [N]
                               int N)
{
    const int n = blockIdx.x * blockDim.x + threadIdx.x;
    if (n >= N) return;
    constexpr int D = 8, K = 5;
    float xv[D];
#pragma unroll
    for (int d = 0; d < D; ++d) xv[d] = x[n * D + d];

    float lp[K];
    float mx = -FLT_BIG;
#pragma unroll
    for (int k = 0; k < K; ++k) {
        float diff[D];
#pragma unroll
        for (int d = 0; d < D; ++d) diff[d] = xv[d] - means[k * D + d];
        float maha = 0.0f, logdet = 0.0f;
#pragma unroll
        for (int e = 0; e < D; ++e) {
            float z = 0.0f;
#pragma unroll
            for (int d = 0; d < D; ++d)
                z = fmaf(diff[d], prec[k * D * D + d * D + e], z);  // nkd,kde->nke
            maha = fmaf(z, z, maha);
            logdet += __logf(prec[k * D * D + e * D + e]);
        }
        lp[k] = -0.5f * ((float)D * LN2PI_F + maha) + logdet + logw[k];
        mx = fmaxf(mx, lp[k]);
    }
    float s = 0.0f;
#pragma unroll
    for (int k = 0; k < K; ++k) s += __expf(lp[k] - mx);
    const float score = -(mx + __logf(s));
    out[n] = (score - vmin[0]) / (vmax[0] - vmin[0]);
}

// ---------------------------------------------------------------------------
// Per-row squared L2 norm: b2[m] = sum_d bank[m][d]^2
// ---------------------------------------------------------------------------
__global__ void rownorm2_kernel(const float* __restrict__ bank, int M, int D,
                                float* __restrict__ b2)
{
    const int m = blockIdx.x * blockDim.x + threadIdx.x;
    if (m >= M) return;
    const float* r = bank + (size_t)m * D;
    float s = 0.0f;
    for (int d = 0; d < D; ++d) s = fmaf(r[d], r[d], s);
    b2[m] = s;
}

// ---------------------------------------------------------------------------
// One-time bank pack: fp32 [M,D] -> bf16 hi/lo [M,KP], zero-padded K.
// ---------------------------------------------------------------------------
template <int D, int KP>
__global__ void pack_bank_kernel(const float* __restrict__ bank,
                                 __bf16* __restrict__ bh,
                                 __bf16* __restrict__ bl, int M)
{
    const long long idx = (long long)blockIdx.x * blockDim.x + threadIdx.x;
    if (idx >= (long long)M * KP) return;
    const int m = (int)(idx / KP);
    const int d = (int)(idx % KP);
    const float f = (d < D) ? bank[(size_t)m * D + d] : 0.0f;
    const __bf16 h = (__bf16)f;
    bh[idx] = h;
    bl[idx] = (__bf16)(f - (float)h);
}

// ---------------------------------------------------------------------------
// 1-NN min distance via WMMA, pre-packed bf16 hi/lo bank.
//   score[n] = sqrt(max(q2[n] + min_m(b2[m] - 2 q.b_m), 1e-12))
// Block = 256 thr (8 waves) owns 16 query rows; each wave processes TWO
// adjacent 16-col bank tiles per iteration so each A-fragment LDS read
// feeds 6 WMMAs.  bf16x3 split: acc += al*bh + ah*bl + ah*bh.
// ---------------------------------------------------------------------------
template <int KP>
__global__ __launch_bounds__(256)
void nn_min_wmma_packed(const float* __restrict__ q,      // [N, D] fp32
                        int D,
                        const __bf16* __restrict__ bkh,   // [M, KP] bf16 hi
                        const __bf16* __restrict__ bkl,   // [M, KP] bf16 lo
                        const float* __restrict__ b2,     // [M]
                        const float* __restrict__ smin,
                        const float* __restrict__ smax,
                        float* __restrict__ out,          // [N] slice
                        int N, int M)
{
    __shared__ __bf16 Ah[16 * KP];
    __shared__ __bf16 Al[16 * KP];
    __shared__ float  red[8][16];

    const int tid    = threadIdx.x;
    const int lane   = tid & 31;
    const int wave   = tid >> 5;
    const int nWaves = blockDim.x >> 5;          // 8
    const int n0     = blockIdx.x * 16;

    for (int idx = tid; idx < 16 * KP; idx += blockDim.x) {
        const int r = idx / KP;
        const int d = idx % KP;
        float f = 0.0f;
        if (d < D && (n0 + r) < N) f = q[(size_t)(n0 + r) * D + d];
        const __bf16 h = (__bf16)f;
        Ah[idx] = h;
        Al[idx] = (__bf16)(f - (float)h);
    }
    __syncthreads();

    // wave32 WMMA fragment indexing: m/n = lane%16; A and B both use contiguous k = 16*(lane/16) + s
    const int row  = lane & 15;
    const int aoff = (lane < 16) ? 0 : 16;
    const int col  = lane & 15;
    const int boff = (lane < 16) ? 0 : 16;

    float rowmin[8];
#pragma unroll
    for (int j = 0; j < 8; ++j) rowmin[j] = FLT_BIG;

    const int nPairs = M >> 5;                   // M assumed multiple of 32
    for (int mp = wave; mp < nPairs; mp += nWaves) {
        const int m0 = mp << 5;                  // two tiles: m0, m0+16
        const __bf16* bh0 = bkh + (size_t)(m0 + col) * KP;
        const __bf16* bl0 = bkl + (size_t)(m0 + col) * KP;
        const __bf16* bh1 = bkh + (size_t)(m0 + 16 + col) * KP;
        const __bf16* bl1 = bkl + (size_t)(m0 + 16 + col) * KP;
        if (mp + nWaves < nPairs) {
            __builtin_prefetch(bkh + (size_t)(m0 + (nWaves << 5) + col) * KP, 0, 1);
            __builtin_prefetch(bkl + (size_t)(m0 + (nWaves << 5) + col) * KP, 0, 1);
        }

        v8f acc0 = {};
        v8f acc1 = {};
#pragma unroll 2
        for (int ks = 0; ks < KP / 32; ++ks) {
            const int kbase = ks * 32;
            // A fragments (hi/lo) from LDS (2x 16B contiguous per frag)
            v16bf ah, al;
            const __bf16* Arh = &Ah[row * KP + kbase];
            const __bf16* Arl = &Al[row * KP + kbase];
#pragma unroll
            for (int i = 0; i < 8; ++i) {
                ah[i]     = Arh[aoff + i];
                ah[i + 8] = Arh[aoff + 8 + i];
                al[i]     = Arl[aoff + i];
                al[i + 8] = Arl[aoff + 8 + i];
            }
            // B fragments: contiguous 32B bf16 loads, no conversion
            v16bf b0h, b0l, b1h, b1l;
#pragma unroll
            for (int i = 0; i < 16; ++i) {
                const int k = kbase + boff + i;
                b0h[i] = bh0[k];
                b0l[i] = bl0[k];
                b1h[i] = bh1[k];
                b1l[i] = bl1[k];
            }
            acc0 = __builtin_amdgcn_wmma_f32_16x16x32_bf16(false, al, false, b0h,
                                                           (short)0, acc0, false, false);
            acc0 = __builtin_amdgcn_wmma_f32_16x16x32_bf16(false, ah, false, b0l,
                                                           (short)0, acc0, false, false);
            acc0 = __builtin_amdgcn_wmma_f32_16x16x32_bf16(false, ah, false, b0h,
                                                           (short)0, acc0, false, false);
            acc1 = __builtin_amdgcn_wmma_f32_16x16x32_bf16(false, al, false, b1h,
                                                           (short)0, acc1, false, false);
            acc1 = __builtin_amdgcn_wmma_f32_16x16x32_bf16(false, ah, false, b1l,
                                                           (short)0, acc1, false, false);
            acc1 = __builtin_amdgcn_wmma_f32_16x16x32_bf16(false, ah, false, b1h,
                                                           (short)0, acc1, false, false);
        }
        const float b2v0 = b2[m0 + col];
        const float b2v1 = b2[m0 + 16 + col];
#pragma unroll
        for (int j = 0; j < 8; ++j) {
            rowmin[j] = fminf(rowmin[j], fmaf(-2.0f, acc0[j], b2v0));
            rowmin[j] = fminf(rowmin[j], fmaf(-2.0f, acc1[j], b2v1));
        }
    }

#pragma unroll
    for (int off = 1; off < 16; off <<= 1) {
#pragma unroll
        for (int j = 0; j < 8; ++j)
            rowmin[j] = fminf(rowmin[j], __shfl_xor(rowmin[j], off, 32));
    }
    if (lane == 0) {
#pragma unroll
        for (int j = 0; j < 8; ++j) red[wave][j] = rowmin[j];
    }
    if (lane == 16) {
#pragma unroll
        for (int j = 0; j < 8; ++j) red[wave][8 + j] = rowmin[j];
    }
    __syncthreads();

    if (tid < 16 && (n0 + tid) < N) {
        float mn = red[0][tid];
        for (int w = 1; w < nWaves; ++w) mn = fminf(mn, red[w][tid]);
        const float* qr = q + (size_t)(n0 + tid) * D;
        float q2 = 0.0f;
        for (int d = 0; d < D; ++d) q2 = fmaf(qr[d], qr[d], q2);
        const float d2 = fmaxf(q2 + mn, 1e-12f);
        const float sc = __builtin_sqrtf(d2);
        out[n0 + tid] = (sc - smin[0]) / (smax[0] - smin[0]);
    }
}

// ---------------------------------------------------------------------------
// Fallback (small workspace): convert B on the fly inside the GEMM loop.
// ---------------------------------------------------------------------------
template <int D, int KP>
__global__ __launch_bounds__(256)
void nn_min_wmma_unpacked(const float* __restrict__ q,
                          const float* __restrict__ bank,
                          const float* __restrict__ b2,
                          const float* __restrict__ smin,
                          const float* __restrict__ smax,
                          float* __restrict__ out,
                          int N, int M)
{
    __shared__ __bf16 Ah[16 * KP];
    __shared__ __bf16 Al[16 * KP];
    __shared__ float  red[8][16];

    const int tid    = threadIdx.x;
    const int lane   = tid & 31;
    const int wave   = tid >> 5;
    const int nWaves = blockDim.x >> 5;
    const int n0     = blockIdx.x * 16;

    for (int idx = tid; idx < 16 * KP; idx += blockDim.x) {
        const int r = idx / KP;
        const int d = idx % KP;
        float f = 0.0f;
        if (d < D && (n0 + r) < N) f = q[(size_t)(n0 + r) * D + d];
        const __bf16 h = (__bf16)f;
        Ah[idx] = h;
        Al[idx] = (__bf16)(f - (float)h);
    }
    __syncthreads();

    const int row  = lane & 15;
    const int aoff = (lane < 16) ? 0 : 16;
    const int col  = lane & 15;
    const int boff = (lane < 16) ? 0 : 16;

    float rowmin[8];
#pragma unroll
    for (int j = 0; j < 8; ++j) rowmin[j] = FLT_BIG;

    const int nTiles = M >> 4;
    for (int mt = wave; mt < nTiles; mt += nWaves) {
        const int m0 = mt << 4;
        const float* brow = bank + (size_t)(m0 + col) * D;
        v8f acc = {};
#pragma unroll 2
        for (int ks = 0; ks < KP / 32; ++ks) {
            const int kbase = ks * 32;
            v16bf ah, al;
            const __bf16* Arh = &Ah[row * KP + kbase];
            const __bf16* Arl = &Al[row * KP + kbase];
#pragma unroll
            for (int i = 0; i < 8; ++i) {
                ah[i]     = Arh[aoff + i];
                ah[i + 8] = Arh[aoff + 8 + i];
                al[i]     = Arl[aoff + i];
                al[i + 8] = Arl[aoff + 8 + i];
            }
            v16bf bh, bl;
#pragma unroll
            for (int i = 0; i < 16; ++i) {
                const int k = kbase + boff + i;
                const float f = (k < D) ? brow[k] : 0.0f;
                const __bf16 h = (__bf16)f;
                bh[i] = h;
                bl[i] = (__bf16)(f - (float)h);
            }
            acc = __builtin_amdgcn_wmma_f32_16x16x32_bf16(false, al, false, bh,
                                                          (short)0, acc, false, false);
            acc = __builtin_amdgcn_wmma_f32_16x16x32_bf16(false, ah, false, bl,
                                                          (short)0, acc, false, false);
            acc = __builtin_amdgcn_wmma_f32_16x16x32_bf16(false, ah, false, bh,
                                                          (short)0, acc, false, false);
        }
        const float b2v = b2[m0 + col];
#pragma unroll
        for (int j = 0; j < 8; ++j)
            rowmin[j] = fminf(rowmin[j], fmaf(-2.0f, acc[j], b2v));
    }

#pragma unroll
    for (int off = 1; off < 16; off <<= 1) {
#pragma unroll
        for (int j = 0; j < 8; ++j)
            rowmin[j] = fminf(rowmin[j], __shfl_xor(rowmin[j], off, 32));
    }
    if (lane == 0) {
#pragma unroll
        for (int j = 0; j < 8; ++j) red[wave][j] = rowmin[j];
    }
    if (lane == 16) {
#pragma unroll
        for (int j = 0; j < 8; ++j) red[wave][8 + j] = rowmin[j];
    }
    __syncthreads();

    if (tid < 16 && (n0 + tid) < N) {
        float mn = red[0][tid];
        for (int w = 1; w < nWaves; ++w) mn = fminf(mn, red[w][tid]);
        const float* qr = q + (size_t)(n0 + tid) * D;
        float q2 = 0.0f;
        for (int d = 0; d < D; ++d) q2 = fmaf(qr[d], qr[d], q2);
        const float d2 = fmaxf(q2 + mn, 1e-12f);
        const float sc = __builtin_sqrtf(d2);
        out[n0 + tid] = (sc - smin[0]) / (smax[0] - smin[0]);
    }
}

// ---------------------------------------------------------------------------
extern "C" void kernel_launch(void* const* d_in, const int* in_sizes, int n_in,
                              void* d_out, int out_size, void* d_ws, size_t ws_size,
                              hipStream_t stream)
{
    const float* velocity   = (const float*)d_in[0];   // [N,8]
    const float* pose       = (const float*)d_in[1];   // [N,34]
    const float* appearance = (const float*)d_in[2];   // [N,512]
    const float* pose_bank  = (const float*)d_in[3];   // [M,34]
    const float* feat_bank  = (const float*)d_in[4];   // [M,512]
    const float* gmm_means  = (const float*)d_in[5];
    const float* gmm_prec   = (const float*)d_in[6];
    const float* gmm_logw   = (const float*)d_in[7];
    const float* vel_min    = (const float*)d_in[8];
    const float* vel_max    = (const float*)d_in[9];
    const float* pose_min   = (const float*)d_in[10];
    const float* pose_max   = (const float*)d_in[11];
    const float* feat_min   = (const float*)d_in[12];
    const float* feat_max   = (const float*)d_in[13];

    const int N = in_sizes[0] / 8;     // 2048
    const int M = in_sizes[3] / 34;    // 65536
    float* out = (float*)d_out;        // [3, N]

    constexpr int KP_POSE = 64;
    constexpr int KP_FEAT = 512;

    // workspace layout
    float*  pose_b2 = (float*)d_ws;                       // [M]
    float*  feat_b2 = pose_b2 + M;                        // [M]
    __bf16* poseh   = (__bf16*)(feat_b2 + M);             // [M*KP_POSE]
    __bf16* posel   = poseh + (size_t)M * KP_POSE;
    __bf16* feath   = posel + (size_t)M * KP_POSE;        // [M*KP_FEAT]
    __bf16* featl   = feath + (size_t)M * KP_FEAT;

    const size_t need = 2ull * M * sizeof(float)
                      + 2ull * M * KP_POSE * sizeof(__bf16)
                      + 2ull * M * KP_FEAT * sizeof(__bf16);

    // bank row norms (always needed)
    rownorm2_kernel<<<(M + 255) / 256, 256, 0, stream>>>(pose_bank, M, 34, pose_b2);
    rownorm2_kernel<<<(M + 255) / 256, 256, 0, stream>>>(feat_bank, M, 512, feat_b2);

    // row 0: velocity GMM
    vel_gmm_kernel<<<(N + 255) / 256, 256, 0, stream>>>(
        velocity, gmm_means, gmm_prec, gmm_logw, vel_min, vel_max, out, N);

    const int nBlocks = (N + 15) / 16;

    if (ws_size >= need) {
        // pre-pack banks to bf16 hi/lo (done every call; still cheap vs GEMM)
        {
            const long long tot = (long long)M * KP_POSE;
            pack_bank_kernel<34, KP_POSE><<<(unsigned)((tot + 255) / 256), 256, 0, stream>>>(
                pose_bank, poseh, posel, M);
        }
        {
            const long long tot = (long long)M * KP_FEAT;
            pack_bank_kernel<512, KP_FEAT><<<(unsigned)((tot + 255) / 256), 256, 0, stream>>>(
                feat_bank, feath, featl, M);
        }
        nn_min_wmma_packed<KP_POSE><<<nBlocks, 256, 0, stream>>>(
            pose, 34, poseh, posel, pose_b2, pose_min, pose_max, out + N, N, M);
        nn_min_wmma_packed<KP_FEAT><<<nBlocks, 256, 0, stream>>>(
            appearance, 512, feath, featl, feat_b2, feat_min, feat_max, out + 2 * N, N, M);
    } else {
        nn_min_wmma_unpacked<34, KP_POSE><<<nBlocks, 256, 0, stream>>>(
            pose, pose_bank, pose_b2, pose_min, pose_max, out + N, N, M);
        nn_min_wmma_unpacked<512, KP_FEAT><<<nBlocks, 256, 0, stream>>>(
            appearance, feat_bank, feat_b2, feat_min, feat_max, out + 2 * N, N, M);
    }
}
